// GraphGeneratorPointNet_68101001445995
// MI455X (gfx1250) — matrix operand order repaired and checked by reading, not compiled, not run on hardware
//
#include <hip/hip_runtime.h>

typedef __attribute__((ext_vector_type(16))) _Float16 v16h;
typedef __attribute__((ext_vector_type(8)))  _Float16 v8h;
typedef __attribute__((ext_vector_type(8)))  float    v8f;
typedef __attribute__((ext_vector_type(4)))  float    v4f;

#define NPTS 512

// ---------------------------------------------------------------------------
// AP[i,k] = (x_i @ W[:FX])[k] + (pos_i @ W[FX:FX+2])[k] + b[k]   (bias folded)
// P [i,k] = pos-part only (no bias) -- used as the per-target subtrahend.
// Factors the first MLP layer out of the N^2 pair loop.
// ---------------------------------------------------------------------------
template<int FX, int H>
__global__ __launch_bounds__(256)
void lin_ap(const float* __restrict__ X, const float* __restrict__ pos,
            const float* __restrict__ W, const float* __restrict__ b,
            float* __restrict__ AP, float* __restrict__ P)
{
    int idx = blockIdx.x * blockDim.x + threadIdx.x;
    if (idx >= NPTS * H) return;
    int i = idx / H, k = idx - i * H;
    float p = pos[2*i] * W[FX*H + k] + pos[2*i+1] * W[(FX+1)*H + k];
    float s = p + b[k];
    for (int t = 0; t < FX; ++t) s += X[i*FX + t] * W[t*H + k];
    P[idx]  = p;
    AP[idx] = s;
}

// ---------------------------------------------------------------------------
// One workgroup per target point i. For all 512 neighbors j (tiles of 16):
//   h_j = relu(AP_j - P_i)            (vectorized build, ds_store_b128, dbl-buf)
//   msg = h_tile @ W2                 (v_wmma_f32_16x16x32_f16, C-chained over K)
//   agg = radius-masked max over j;   out = relu(agg + b2)
// Each wave owns 16 output columns; its W2 slice lives in registers as
// pre-converted f16 B-fragments for the whole kernel.
// ---------------------------------------------------------------------------
template<int H, int H2>
__global__ __launch_bounds__(H2*2)
void point_conv_wmma(const float* __restrict__ pos,
                     const float* __restrict__ AP, const float* __restrict__ P,
                     const float* __restrict__ W2, const float* __restrict__ b2,
                     float* __restrict__ out, float r2)
{
    static_assert(H == H2, "tile-build vectorization assumes H == H2");
    constexpr int KS = H / 32;            // K-steps of 32
    const int i    = blockIdx.x;
    const int tid  = threadIdx.x;
    const int wave = tid >> 5;
    const int lane = tid & 31;
    const int col  = lane & 15;           // fragment column / A row
    const int hi   = lane >> 4;           // lane half

    __shared__ __align__(16) float    sPi[H];
    __shared__ __align__(16) _Float16 sH[2][16 * H];   // double-buffered LHS tiles

    for (int k = tid; k < H; k += blockDim.x) sPi[k] = P[i*H + k];
    __syncthreads();
    const float pix = pos[2*i], piy = pos[2*i+1];

    // Preload this wave's 16 columns of W2 as B-fragments (contiguous K order, same as the A reads:
    // lanes 0-15 hold K = 0..15 of the step, lanes 16-31 hold K = 16..31).
    v16h bfrag[KS];
    const int bcol = wave * 16 + col;
    #pragma unroll
    for (int s = 0; s < KS; ++s) {
        #pragma unroll
        for (int p = 0; p < 8; ++p) {
            int k0 = s*32 + hi*16 + 2*p;
            bfrag[s][2*p]   = (_Float16)W2[(k0    )*H2 + bcol];
            bfrag[s][2*p+1] = (_Float16)W2[(k0 + 1)*H2 + bcol];
        }
    }

    // Each thread owns exactly 8 contiguous elements of the 16 x H tile.
    const int e0 = tid * 8;
    const int jl = e0 / H;                // row (neighbor) within tile
    const int k0 = e0 - jl * H;           // column start (multiple of 8)

    float colmax = -3.0e38f;              // ~ -inf; final relu handles "no neighbor"

    for (int tile = 0; tile < NPTS/16; ++tile) {
        _Float16* sHb = sH[tile & 1];

        // Radius mask for this tile's 16 rows (bits 0..15 of wave ballot).
        int   jm = tile*16 + col;
        float dx = pos[2*jm] - pix, dy = pos[2*jm+1] - piy;
        bool  pred = (hi == 0) && (dx*dx + dy*dy <= r2) && (jm != i);
        unsigned mask = (unsigned)(__ballot(pred) & 0xFFFFu);

        // Build h = relu(AP_j - P_i): 2x float4 loads -> packed f16 -> b128 store.
        {
            const float* apRow = AP + (size_t)(tile*16 + jl) * H + k0;
            v4f a0 = *(const v4f*)(apRow);
            v4f a1 = *(const v4f*)(apRow + 4);
            v4f p0 = *(const v4f*)(&sPi[k0]);
            v4f p1 = *(const v4f*)(&sPi[k0 + 4]);
            v8h hv;
            #pragma unroll
            for (int q = 0; q < 4; ++q) {
                hv[q]     = (_Float16)fmaxf(a0[q] - p0[q], 0.f);
                hv[4 + q] = (_Float16)fmaxf(a1[q] - p1[q], 0.f);
            }
            *(v8h*)(&sHb[e0]) = hv;
        }
        // Pull the next tile's AP slab toward the WGP while WMMAs run.
        if (tile + 1 < NPTS/16)
            __builtin_prefetch(AP + (size_t)(tile + 1) * 16 * H + e0, 0, 1);

        __syncthreads();   // build(t) done; wmma(t) may start. build(t+1) uses other buffer.

        // 16x16 tile GEMM: accumulate over K via C-chaining.
        v8f acc = {};
        #pragma unroll
        for (int s = 0; s < KS; ++s) {
            // A layout: one contiguous 16-half run per lane (K = 16*hi..16*hi+15) -> 2x ds_load_b128.
            const _Float16* rp = &sHb[col*H + s*32 + hi*16];
            v8h alo = *(const v8h*)(rp);
            v8h ahi = *(const v8h*)(rp + 8);
            v16h a = __builtin_shufflevector(alo, ahi,
                       0,1,2,3,4,5,6,7,8,9,10,11,12,13,14,15);
            acc = __builtin_amdgcn_wmma_f32_16x16x32_f16(
                      false, a, false, bfrag[s], (short)0, acc, false, false);
        }

        // Masked max over the 8 neighbor-rows this lane holds.
        #pragma unroll
        for (int v = 0; v < 8; ++v) {
            int row = v + hi*8;           // C layout: row = v + 8*(lane>=16)
            if ((mask >> row) & 1u) colmax = fmaxf(colmax, acc[v]);
        }
    }

    // Merge rows 0-7 (lanes 0-15) with rows 8-15 (lanes 16-31): same column.
    colmax = fmaxf(colmax, __shfl_xor(colmax, 16, 32));
    if (hi == 0) {
        int oc = wave*16 + col;
        out[i*H2 + oc] = fmaxf(colmax + b2[oc], 0.f);   // wrapper relu
    }
}

// ---------------------------------------------------------------------------
// Mean-pool both clouds, per-cloud Lin(256,128)+relu, concat, then the head:
// Lin(256,128)+relu -> Lin(128,64)+relu -> Lin(64,2); out = [gen + c, gen].
// ---------------------------------------------------------------------------
__global__ __launch_bounds__(256)
void pool_head(const float* __restrict__ x3o, const float* __restrict__ x3m,
               const float* __restrict__ l3w, const float* __restrict__ l3b,
               const float* __restrict__ l6w, const float* __restrict__ l6b,
               const float* __restrict__ l4w, const float* __restrict__ l4b,
               const float* __restrict__ l5w, const float* __restrict__ l5b,
               const float* __restrict__ cvec, float* __restrict__ out)
{
    __shared__ float g[2][256];
    __shared__ float prior[256];
    __shared__ float h1[128];
    __shared__ float h2[64];
    int t = threadIdx.x;

    for (int c = 0; c < 2; ++c) {
        const float* x3 = c ? x3m : x3o;
        float s = 0.f;
        for (int r = 0; r < NPTS; ++r) s += x3[r*256 + t];
        g[c][t] = s * (1.f / NPTS);
    }
    __syncthreads();
    {
        int c = t >> 7, f = t & 127;
        float s = l3b[f];
        for (int k = 0; k < 256; ++k) s += g[c][k] * l3w[k*128 + f];
        prior[c*128 + f] = fmaxf(s, 0.f);
    }
    __syncthreads();
    if (t < 128) {
        float s = l6b[t];
        for (int k = 0; k < 256; ++k) s += prior[k] * l6w[k*128 + t];
        h1[t] = fmaxf(s, 0.f);
    }
    __syncthreads();
    if (t < 64) {
        float s = l4b[t];
        for (int k = 0; k < 128; ++k) s += h1[k] * l4w[k*64 + t];
        h2[t] = fmaxf(s, 0.f);
    }
    __syncthreads();
    if (t < 2) {
        float s = l5b[t];
        for (int k = 0; k < 64; ++k) s += h2[k] * l5w[k*2 + t];
        out[t]     = s + cvec[t];   // gen + c
        out[2 + t] = s;             // gen
    }
}

// ---------------------------------------------------------------------------
extern "C" void kernel_launch(void* const* d_in, const int* /*in_sizes*/, int /*n_in*/,
                              void* d_out, int /*out_size*/, void* d_ws, size_t /*ws_size*/,
                              hipStream_t stream)
{
    const float* o   = (const float*)d_in[0];
    const float* m   = (const float*)d_in[1];
    const float* cv  = (const float*)d_in[2];
    const float* w1a = (const float*)d_in[3];  const float* b1a = (const float*)d_in[4];
    const float* w1b = (const float*)d_in[5];  const float* b1b = (const float*)d_in[6];
    const float* w2a = (const float*)d_in[7];  const float* b2a = (const float*)d_in[8];
    const float* w2b = (const float*)d_in[9];  const float* b2b = (const float*)d_in[10];
    const float* w3a = (const float*)d_in[11]; const float* b3a = (const float*)d_in[12];
    const float* w3b = (const float*)d_in[13]; const float* b3b = (const float*)d_in[14];
    const float* l3w = (const float*)d_in[15]; const float* l3b = (const float*)d_in[16];
    const float* l6w = (const float*)d_in[17]; const float* l6b = (const float*)d_in[18];
    const float* l4w = (const float*)d_in[19]; const float* l4b = (const float*)d_in[20];
    const float* l5w = (const float*)d_in[21]; const float* l5b = (const float*)d_in[22];

    float* ws = (float*)d_ws;
    const size_t PC = (size_t)NPTS * (64*3 + 128*3 + 256*3);  // per-cloud floats (~2.75MB)

    float* x3c[2];
    for (int c = 0; c < 2; ++c) {
        const float* pos = c ? m : o;
        float* base = ws + (size_t)c * PC;
        float* P1  = base;
        float* P1b = P1  + NPTS*64;
        float* x1  = P1b + NPTS*64;
        float* AP2 = x1  + NPTS*64;
        float* P2  = AP2 + NPTS*128;
        float* x2  = P2  + NPTS*128;
        float* AP3 = x2  + NPTS*128;
        float* P3  = AP3 + NPTS*256;
        float* x3  = P3  + NPTS*256;
        x3c[c] = x3;

        lin_ap<0,64><<<(NPTS*64 + 255)/256, 256, 0, stream>>>(nullptr, pos, w1a, b1a, P1, P1b);
        point_conv_wmma<64,64><<<NPTS, 128, 0, stream>>>(pos, P1, P1b, w1b, b1b, x1, 0.04f);

        lin_ap<64,128><<<(NPTS*128 + 255)/256, 256, 0, stream>>>(x1, pos, w2a, b2a, AP2, P2);
        point_conv_wmma<128,128><<<NPTS, 256, 0, stream>>>(pos, AP2, P2, w2b, b2b, x2, 0.16f);

        lin_ap<128,256><<<(NPTS*256 + 255)/256, 256, 0, stream>>>(x2, pos, w3a, b3a, AP3, P3);
        point_conv_wmma<256,256><<<NPTS, 512, 0, stream>>>(pos, AP3, P3, w3b, b3b, x3, 0.36f);
    }

    pool_head<<<1, 256, 0, stream>>>(x3c[0], x3c[1], l3w, l3b, l6w, l6b,
                                     l4w, l4b, l5w, l5b, cv, (float*)d_out);
}
